// Longformer_5205500363376
// MI455X (gfx1250) — matrix operand order repaired and checked by reading, not compiled, not run on hardware
//
#include <hip/hip_runtime.h>
#include <hip/hip_bf16.h>

#define DEV __device__ __forceinline__

typedef __attribute__((ext_vector_type(16))) _Float16 v16h;
typedef __attribute__((ext_vector_type(8)))  _Float16 h8;
typedef __attribute__((ext_vector_type(8)))  float    v8f;
typedef int v4i __attribute__((vector_size(4 * sizeof(int))));

#if defined(__has_builtin)
#if __has_builtin(__builtin_amdgcn_global_load_async_to_lds_b128) && \
    __has_builtin(__builtin_amdgcn_s_wait_asynccnt)
#define USE_ASYNC_LDS 1
#endif
#endif
#ifndef USE_ASYNC_LDS
#define USE_ASYNC_LDS 0
#endif

#if USE_ASYNC_LDS
#define AS1P(p) ((__attribute__((address_space(1))) v4i*)(p))
#define AS3P(p) ((__attribute__((address_space(3))) v4i*)(p))
#endif

namespace cfg {
constexpr int B   = 2;
constexpr int S   = 4096;
constexpr int D   = 768;
constexpr int H   = 12;
constexpr int HD  = 64;
constexpr int W   = 256;
constexpr int G   = 64;
constexpr int DFF = 3072;
constexpr int NB  = S / W;     // 16
constexpr int M   = B * S;     // 8192
}

// ---------------------------------------------------------------------------
// Fragment loaders (wave32, 16x16x32 f16 WMMA)
// A-frag (16x32, M x K): lane holds row M=lane%16; halves 0..7 -> K = kb..kb+7,
// halves 8..15 -> K = kb+16..kb+23 with kb = (lane&16)?8:0.
// B-frag (32x16, K x N): lane holds col N=lane%16; same K order as A-frag
// (halves 0..7 -> K = kb..kb+7, 8..15 -> kb+16..kb+23, kb = (lane&16)?8:0).
// C/D f32 16x16: vgpr r: lanes0-15 (M=r, N=lane), lanes16-31 (M=r+8, N=lane-16)
// ---------------------------------------------------------------------------
DEV v16h load2(const _Float16* p0, const _Float16* p1) {
  union { v16h v; h8 g[2]; } u;
  u.g[0] = *(const h8*)p0;
  u.g[1] = *(const h8*)p1;
  return u.v;
}
DEV v16h load_afrag(const _Float16* row, int k0, int lane) {
  int kb = k0 + ((lane & 16) ? 8 : 0);
  return load2(row + kb, row + kb + 16);
}
DEV v16h load_bfrag(const _Float16* row, int k0, int lane) {
  int kb = k0 + ((lane & 16) ? 8 : 0);
  return load2(row + kb, row + kb + 16);
}
DEV v8f wmma_f16(v16h a, v16h b, v8f c) {
  return __builtin_amdgcn_wmma_f32_16x16x32_f16(false, a, false, b, (short)0, c,
                                                false, false);
}
DEV float gelu_exact(float x) {
  return 0.5f * x * (1.0f + erff(x * 0.7071067811865476f));
}

// ---------------------------------------------------------------------------
// Elementwise conversion kernels
// ---------------------------------------------------------------------------
__global__ void f32_to_f16_k(const float* __restrict__ in,
                             _Float16* __restrict__ out, size_t n) {
  size_t i = (size_t)blockIdx.x * 256 + threadIdx.x;
  if (i < n) out[i] = (_Float16)in[i];
}

// in: f32 [K][N] row-major  ->  out: f16 [N][K]
__global__ void transpose_f16_k(const float* __restrict__ in,
                                _Float16* __restrict__ out, int K, int N) {
  size_t i = (size_t)blockIdx.x * 256 + threadIdx.x;
  if (i >= (size_t)K * N) return;
  int kk = (int)(i / N);
  int nn = (int)(i % N);
  out[(size_t)nn * K + kk] = (_Float16)in[i];
}

// ---------------------------------------------------------------------------
// Generic WMMA GEMM: C[M,N] = A[M,K] * Bt[N,K]^T + bias
// block = 128 (4 waves); block tile 64x64; each wave computes a 16x64 strip.
// A/B K-tiles (64x32 f16 each) are staged in LDS, double buffered; staging
// uses the CDNA5 async global->LDS DMA path when available (ASYNCcnt),
// overlapping the next tile's copy with the current tile's WMMAs.
// mode 0: f32 [M,N]   mode 1: f16 [M,N]
// mode 2: f16 heads   [B,H,S,HD]   (M = B*S, N = D)
// mode 3: f16 headsT  [B,H,HD,S]
// act  1: exact GELU
// ---------------------------------------------------------------------------
__global__ __launch_bounds__(128) void gemm_wmma_k(
    const _Float16* __restrict__ A, const _Float16* __restrict__ Bt,
    const float* __restrict__ bias, void* __restrict__ Cout,
    int Mm, int Nn, int Kk, int mode, int act) {
  using namespace cfg;
  __shared__ __align__(16) _Float16 lA[2][64][32];
  __shared__ __align__(16) _Float16 lB[2][64][32];

  const int tid   = threadIdx.x;
  const int lane  = tid & 31;
  const int wave  = tid >> 5;
  const int lhalf = lane & 15;
  const int row0  = blockIdx.x * 64;
  const int col0  = blockIdx.y * 64;

  // Loader mapping: a 64x32 f16 tile = 256 16-byte chunks; 2 chunks/thread.
  // chunk c: row = c>>2, koff = (c&3)*8 halves.
  const int c0  = tid * 2;
  const int lr0 = c0 >> 2;
  const int lk0 = (c0 & 3) * 8;
  const int lr1 = (c0 + 1) >> 2;
  const int lk1 = ((c0 + 1) & 3) * 8;

  const _Float16* Ag0 = A + (size_t)(row0 + lr0) * Kk + lk0;
  const _Float16* Ag1 = A + (size_t)(row0 + lr1) * Kk + lk1;
  const _Float16* Bg0 = Bt + (size_t)(col0 + lr0) * Kk + lk0;
  const _Float16* Bg1 = Bt + (size_t)(col0 + lr1) * Kk + lk1;

  auto stage = [&](int buf, int k0) {
#if USE_ASYNC_LDS
    __builtin_amdgcn_global_load_async_to_lds_b128(
        AS1P(Ag0 + k0), AS3P(&lA[buf][lr0][lk0]), 0, 0);
    __builtin_amdgcn_global_load_async_to_lds_b128(
        AS1P(Ag1 + k0), AS3P(&lA[buf][lr1][lk1]), 0, 0);
    __builtin_amdgcn_global_load_async_to_lds_b128(
        AS1P(Bg0 + k0), AS3P(&lB[buf][lr0][lk0]), 0, 0);
    __builtin_amdgcn_global_load_async_to_lds_b128(
        AS1P(Bg1 + k0), AS3P(&lB[buf][lr1][lk1]), 0, 0);
#else
    h8 a0 = *(const h8*)(Ag0 + k0);
    h8 a1 = *(const h8*)(Ag1 + k0);
    h8 b0 = *(const h8*)(Bg0 + k0);
    h8 b1 = *(const h8*)(Bg1 + k0);
    *(h8*)&lA[buf][lr0][lk0] = a0;
    *(h8*)&lA[buf][lr1][lk1] = a1;
    *(h8*)&lB[buf][lr0][lk0] = b0;
    *(h8*)&lB[buf][lr1][lk1] = b1;
#endif
  };

  const int KT = Kk / 32;
  stage(0, 0);
#if USE_ASYNC_LDS
  __builtin_amdgcn_s_wait_asynccnt(0);
#endif
  __syncthreads();

  v8f acc[4] = {};
  for (int kt = 0; kt < KT; ++kt) {
    const int cur = kt & 1;
    if (kt + 1 < KT) stage(cur ^ 1, (kt + 1) * 32);

    const v16h af = load_afrag(&lA[cur][wave * 16 + lhalf][0], 0, lane);
#pragma unroll
    for (int j = 0; j < 4; ++j) {
      const v16h bf = load_bfrag(&lB[cur][j * 16 + lhalf][0], 0, lane);
      acc[j] = wmma_f16(af, bf, acc[j]);
    }

    if (kt + 1 < KT) {
#if USE_ASYNC_LDS
      __builtin_amdgcn_s_wait_asynccnt(0);
#endif
      __syncthreads();
    }
  }

  const int rbase = row0 + wave * 16 + ((lane & 16) ? 8 : 0);
#pragma unroll
  for (int j = 0; j < 4; ++j) {
    const int n  = col0 + j * 16 + lhalf;
    const float bv = bias ? bias[n] : 0.0f;
#pragma unroll
    for (int r = 0; r < 8; ++r) {
      const int m = rbase + r;
      float v = acc[j][r] + bv;
      if (act == 1) v = gelu_exact(v);
      if (mode == 0) {
        ((float*)Cout)[(size_t)m * Nn + n] = v;
      } else if (mode == 1) {
        ((_Float16*)Cout)[(size_t)m * Nn + n] = (_Float16)v;
      } else {
        const int bb = m >> 12;          // S == 4096
        const int s  = m & (S - 1);
        const int hh = n >> 6;           // HD == 64
        const int hd = n & (HD - 1);
        if (mode == 2)
          ((_Float16*)Cout)[(((size_t)(bb * H + hh) * S + s) * HD) + hd] =
              (_Float16)v;
        else
          ((_Float16*)Cout)[(((size_t)(bb * H + hh) * HD + hd) * S) + s] =
              (_Float16)v;
      }
    }
  }
}

// ---------------------------------------------------------------------------
// Sliding-window (band + global-key) attention, flash-style online softmax.
// grid = B*H*NB*(W/64); block = 128 (4 waves); wave handles 16 query rows.
// 26 chunks of 32 keys: chunks 0..23 band keys (ki = (n-1)*W + c*32 ..),
// chunks 24..25 the G=64 global keys (attending k[:, :G], v[:, :G]).
// attn output layout: f16 [B, S, D]  (== [B,S,H,HD]).
// ---------------------------------------------------------------------------
__global__ __launch_bounds__(128) void band_attn_k(
    const _Float16* __restrict__ q, const _Float16* __restrict__ k,
    const _Float16* __restrict__ vT, const int* __restrict__ amask,
    _Float16* __restrict__ attn) {
  using namespace cfg;
  __shared__ __align__(16) _Float16 pl[4][16][32];

  const int lane  = threadIdx.x & 31;
  const int wave  = threadIdx.x >> 5;
  const int lhalf = lane & 15;
  const int hi    = (lane & 16) ? 8 : 0;

  int bid = blockIdx.x;
  const int qt   = bid & 3;  bid >>= 2;
  const int nblk = bid & 15; bid >>= 4;
  const int hh   = bid % H;
  const int bb   = bid / H;

  const int    qrow0 = qt * 64 + wave * 16;         // local in [0, W)
  const size_t bhS   = (size_t)(bb * H + hh) * S;

  const _Float16* qrow = q + (bhS + (size_t)(nblk * W + qrow0 + lhalf)) * HD;
  const v16h aq0 = load_afrag(qrow, 0, lane);
  const v16h aq1 = load_afrag(qrow, 32, lane);

  float mrow[8], lrow[8];
  v8f o[4] = {};
#pragma unroll
  for (int r = 0; r < 8; ++r) { mrow[r] = -3.0e38f; lrow[r] = 0.0f; }

  const float scale   = 0.125f;  // 1/sqrt(HD)
  const int   sq_base = nblk * W + qrow0 + hi;

  for (int c = 0; c < 26; ++c) {
    const bool isg     = (c >= 24);
    const int  keybase = isg ? (c - 24) * 32 : (nblk - 1) * W + c * 32;

    v8f sc[2];
#pragma unroll
    for (int t = 0; t < 2; ++t) {
      const int kiN  = keybase + t * 16 + lhalf;
      const int krow = min(max(kiN, 0), S - 1);
      const _Float16* kr = k + (bhS + krow) * HD;
      v8f s = {};
      s = wmma_f16(aq0, load_bfrag(kr, 0, lane), s);
      s = wmma_f16(aq1, load_bfrag(kr, 32, lane), s);
#pragma unroll
      for (int r = 0; r < 8; ++r) {
        bool ok;
        if (isg) {
          ok = amask[bb * S + kiN] > 0;
        } else {
          const int sq  = sq_base + r;
          const int rel = kiN - sq;
          ok = (rel >= -W) && (rel <= W) && (kiN >= G) && (kiN < S) &&
               (amask[bb * S + min(max(kiN, 0), S - 1)] > 0);
        }
        s[r] = ok ? s[r] * scale : -1.0e9f;
      }
      sc[t] = s;
    }

    // online softmax update (row reductions across the 16-lane half)
#pragma unroll
    for (int r = 0; r < 8; ++r) {
      float mx = fmaxf(sc[0][r], sc[1][r]);
#pragma unroll
      for (int d = 1; d <= 8; d <<= 1) mx = fmaxf(mx, __shfl_xor(mx, d, 32));
      const float nm    = fmaxf(mrow[r], mx);
      const float alpha = __expf(mrow[r] - nm);
      const float p0    = __expf(sc[0][r] - nm);
      const float p1    = __expf(sc[1][r] - nm);
      float psum = p0 + p1;
#pragma unroll
      for (int d = 1; d <= 8; d <<= 1) psum += __shfl_xor(psum, d, 32);
      lrow[r] = lrow[r] * alpha + psum;
      mrow[r] = nm;
#pragma unroll
      for (int j = 0; j < 4; ++j) o[j][r] *= alpha;
      pl[wave][hi + r][lhalf]      = (_Float16)p0;   // same-wave LDS: in-order
      pl[wave][hi + r][16 + lhalf] = (_Float16)p1;
    }

    // reshape P (16x32) into an A-fragment via per-wave LDS tile
    const int kb = (lane & 16) ? 8 : 0;
    const v16h pa = load2(&pl[wave][lhalf][kb], &pl[wave][lhalf][kb + 16]);

    // P @ V chunk: V rows are keys, transposed layout vT[b,h,hd,s]
#pragma unroll
    for (int j = 0; j < 4; ++j) {
      const int hd = j * 16 + lhalf;
      const _Float16* vrow = vT + ((size_t)(bb * H + hh) * HD + hd) * S;
      const int kbB = keybase + ((lane & 16) ? 8 : 0);
      const int g0  = min(max(kbB, 0), S - 8);       // masked cols have P==0
      const int g1  = min(max(kbB + 16, 0), S - 8);
      o[j] = wmma_f16(pa, load2(vrow + g0, vrow + g1), o[j]);
    }
  }

#pragma unroll
  for (int j = 0; j < 4; ++j) {
    const int hd = j * 16 + lhalf;
#pragma unroll
    for (int r = 0; r < 8; ++r) {
      const int   srow = sq_base + r;
      const float lr   = lrow[r];
      const float ov   = o[j][r] / (lr > 0.0f ? lr : 1.0f);
      attn[((size_t)bb * S + srow) * D + hh * HD + hd] = (_Float16)ov;
    }
  }
}

// ---------------------------------------------------------------------------
// Global attention for the first G=64 queries: full softmax over all S keys.
// grid = B*H; block = 128 (4 waves x 16 rows = 64 = G rows).
// Overwrites attn rows s < G (launched after band_attn_k).
// ---------------------------------------------------------------------------
__global__ __launch_bounds__(128) void global_attn_k(
    const _Float16* __restrict__ qg, const _Float16* __restrict__ kg,
    const _Float16* __restrict__ vgT, const int* __restrict__ amask,
    _Float16* __restrict__ attn) {
  using namespace cfg;
  __shared__ __align__(16) _Float16 pl[4][16][32];

  const int lane  = threadIdx.x & 31;
  const int wave  = threadIdx.x >> 5;
  const int lhalf = lane & 15;
  const int hi    = (lane & 16) ? 8 : 0;
  const int bb    = blockIdx.x / H;
  const int hh    = blockIdx.x % H;

  const size_t bhS = (size_t)(bb * H + hh) * S;
  const _Float16* qrow = qg + (bhS + (size_t)(wave * 16 + lhalf)) * HD;
  const v16h aq0 = load_afrag(qrow, 0, lane);
  const v16h aq1 = load_afrag(qrow, 32, lane);

  float mrow[8], lrow[8];
  v8f o[4] = {};
#pragma unroll
  for (int r = 0; r < 8; ++r) { mrow[r] = -3.0e38f; lrow[r] = 0.0f; }
  const float scale = 0.125f;

  for (int c = 0; c < S / 32; ++c) {
    const int keybase = c * 32;
    v8f sc[2];
#pragma unroll
    for (int t = 0; t < 2; ++t) {
      const int kiN = keybase + t * 16 + lhalf;
      const _Float16* kr = kg + (bhS + kiN) * HD;
      v8f s = {};
      s = wmma_f16(aq0, load_bfrag(kr, 0, lane), s);
      s = wmma_f16(aq1, load_bfrag(kr, 32, lane), s);
      const bool ok = amask[bb * S + kiN] > 0;
#pragma unroll
      for (int r = 0; r < 8; ++r) s[r] = ok ? s[r] * scale : -1.0e9f;
      sc[t] = s;
    }

#pragma unroll
    for (int r = 0; r < 8; ++r) {
      float mx = fmaxf(sc[0][r], sc[1][r]);
#pragma unroll
      for (int d = 1; d <= 8; d <<= 1) mx = fmaxf(mx, __shfl_xor(mx, d, 32));
      const float nm    = fmaxf(mrow[r], mx);
      const float alpha = __expf(mrow[r] - nm);
      const float p0    = __expf(sc[0][r] - nm);
      const float p1    = __expf(sc[1][r] - nm);
      float psum = p0 + p1;
#pragma unroll
      for (int d = 1; d <= 8; d <<= 1) psum += __shfl_xor(psum, d, 32);
      lrow[r] = lrow[r] * alpha + psum;
      mrow[r] = nm;
#pragma unroll
      for (int j = 0; j < 4; ++j) o[j][r] *= alpha;
      pl[wave][hi + r][lhalf]      = (_Float16)p0;
      pl[wave][hi + r][16 + lhalf] = (_Float16)p1;
    }

    const int kb = (lane & 16) ? 8 : 0;
    const v16h pa = load2(&pl[wave][lhalf][kb], &pl[wave][lhalf][kb + 16]);

#pragma unroll
    for (int j = 0; j < 4; ++j) {
      const int hd = j * 16 + lhalf;
      const _Float16* vrow = vgT + ((size_t)(bb * H + hh) * HD + hd) * S;
      const int kbB = keybase + ((lane & 16) ? 8 : 0);
      o[j] = wmma_f16(pa, load2(vrow + kbB, vrow + kbB + 16), o[j]);
    }
  }

#pragma unroll
  for (int j = 0; j < 4; ++j) {
    const int hd = j * 16 + lhalf;
#pragma unroll
    for (int r = 0; r < 8; ++r) {
      const int   srow = wave * 16 + hi + r;          // in [0, G)
      const float lr   = lrow[r];
      const float ov   = o[j][r] / (lr > 0.0f ? lr : 1.0f);
      attn[((size_t)bb * S + srow) * D + hh * HD + hd] = (_Float16)ov;
    }
  }
}

// ---------------------------------------------------------------------------
// out = LayerNorm(a + b) * g + beta; emits f32 and (optionally) f16 copies.
// one wave per token (D = 768 = 24 * 32); block 128 -> 4 tokens.
// ---------------------------------------------------------------------------
__global__ __launch_bounds__(128) void add_ln_k(
    const float* __restrict__ a, const float* __restrict__ bx,
    const float* __restrict__ g, const float* __restrict__ beta,
    float* __restrict__ out32, _Float16* __restrict__ out16, int ntok) {
  using namespace cfg;
  const int lane = threadIdx.x & 31;
  const int wave = threadIdx.x >> 5;
  const int tok  = blockIdx.x * 4 + wave;
  if (tok >= ntok) return;
  const size_t base = (size_t)tok * D;

  float vv[24];
  float s = 0.0f, s2 = 0.0f;
#pragma unroll
  for (int t = 0; t < 24; ++t) {
    const int i = lane + t * 32;
    const float v = a[base + i] + bx[base + i];
    vv[t] = v;
    s += v;
    s2 += v * v;
  }
#pragma unroll
  for (int d = 1; d <= 16; d <<= 1) {
    s  += __shfl_xor(s, d, 32);
    s2 += __shfl_xor(s2, d, 32);
  }
  const float mu  = s * (1.0f / D);
  const float var = s2 * (1.0f / D) - mu * mu;
  const float rs  = rsqrtf(var + 1e-5f);
#pragma unroll
  for (int t = 0; t < 24; ++t) {
    const int i = lane + t * 32;
    const float v = (vv[t] - mu) * rs * g[i] + beta[i];
    out32[base + i] = v;
    if (out16) out16[base + i] = (_Float16)v;
  }
}

// ---------------------------------------------------------------------------
// Host orchestration
// ---------------------------------------------------------------------------
extern "C" void kernel_launch(void* const* d_in, const int* in_sizes, int n_in,
                              void* d_out, int out_size, void* d_ws,
                              size_t ws_size, hipStream_t stream) {
  using namespace cfg;
  (void)in_sizes; (void)n_in; (void)out_size; (void)ws_size;

  const float* x     = (const float*)d_in[0];
  const float* Wq    = (const float*)d_in[1];
  const float* bq    = (const float*)d_in[2];
  const float* Wk    = (const float*)d_in[3];
  const float* bk    = (const float*)d_in[4];
  const float* Wv    = (const float*)d_in[5];
  const float* bv    = (const float*)d_in[6];
  const float* Wqg   = (const float*)d_in[7];
  const float* bqg   = (const float*)d_in[8];
  const float* Wkg   = (const float*)d_in[9];
  const float* bkg   = (const float*)d_in[10];
  const float* Wvg   = (const float*)d_in[11];
  const float* bvg   = (const float*)d_in[12];
  const float* Wo    = (const float*)d_in[13];
  const float* bo    = (const float*)d_in[14];
  const float* ln1g  = (const float*)d_in[15];
  const float* ln1b  = (const float*)d_in[16];
  const float* Wf1   = (const float*)d_in[17];
  const float* bf1   = (const float*)d_in[18];
  const float* Wf2   = (const float*)d_in[19];
  const float* bf2   = (const float*)d_in[20];
  const float* ln2g  = (const float*)d_in[21];
  const float* ln2b  = (const float*)d_in[22];
  const int*   amask = (const int*)d_in[23];

  char*  base = (char*)d_ws;
  size_t off  = 0;
  auto alloc = [&](size_t elems, size_t esz) -> void* {
    void* p = base + off;
    off += (elems * esz + 255) & ~(size_t)255;
    return p;
  };

  _Float16* xb    = (_Float16*)alloc((size_t)M * D, 2);
  _Float16* wqt   = (_Float16*)alloc((size_t)D * D, 2);
  _Float16* wkt   = (_Float16*)alloc((size_t)D * D, 2);
  _Float16* wvt   = (_Float16*)alloc((size_t)D * D, 2);
  _Float16* wqgt  = (_Float16*)alloc((size_t)D * D, 2);
  _Float16* wkgt  = (_Float16*)alloc((size_t)D * D, 2);
  _Float16* wvgt  = (_Float16*)alloc((size_t)D * D, 2);
  _Float16* wot   = (_Float16*)alloc((size_t)D * D, 2);
  _Float16* wf1t  = (_Float16*)alloc((size_t)D * DFF, 2);
  _Float16* wf2t  = (_Float16*)alloc((size_t)DFF * D, 2);
  _Float16* qh    = (_Float16*)alloc((size_t)M * D, 2);   // [B,H,S,HD]
  _Float16* kh    = (_Float16*)alloc((size_t)M * D, 2);   // [B,H,S,HD]
  _Float16* vTh   = (_Float16*)alloc((size_t)M * D, 2);   // [B,H,HD,S]
  _Float16* qgh   = (_Float16*)alloc((size_t)M * D, 2);   // [B,H,S,HD]
  _Float16* kgh   = (_Float16*)alloc((size_t)M * D, 2);   // [B,H,S,HD]
  _Float16* vgTh  = (_Float16*)alloc((size_t)M * D, 2);   // [B,H,HD,S]
  _Float16* attn  = (_Float16*)alloc((size_t)M * D, 2);   // [B,S,D]
  float*    tmp   = (float*)alloc((size_t)M * D, 4);
  float*    h1    = (float*)alloc((size_t)M * D, 4);
  _Float16* h1h   = (_Float16*)alloc((size_t)M * D, 2);
  _Float16* ff    = (_Float16*)alloc((size_t)M * DFF, 2);
  float*    tmp2  = (float*)alloc((size_t)M * D, 4);

  // --- precision conversion + weight transposition ---
  {
    size_t n = (size_t)M * D;
    f32_to_f16_k<<<dim3((unsigned)((n + 255) / 256)), 256, 0, stream>>>(x, xb, n);
  }
  const unsigned gdd = (unsigned)(((size_t)D * D + 255) / 256);
  transpose_f16_k<<<dim3(gdd), 256, 0, stream>>>(Wq,  wqt,  D, D);
  transpose_f16_k<<<dim3(gdd), 256, 0, stream>>>(Wk,  wkt,  D, D);
  transpose_f16_k<<<dim3(gdd), 256, 0, stream>>>(Wv,  wvt,  D, D);
  transpose_f16_k<<<dim3(gdd), 256, 0, stream>>>(Wqg, wqgt, D, D);
  transpose_f16_k<<<dim3(gdd), 256, 0, stream>>>(Wkg, wkgt, D, D);
  transpose_f16_k<<<dim3(gdd), 256, 0, stream>>>(Wvg, wvgt, D, D);
  transpose_f16_k<<<dim3(gdd), 256, 0, stream>>>(Wo,  wot,  D, D);
  const unsigned gdf = (unsigned)(((size_t)D * DFF + 255) / 256);
  transpose_f16_k<<<dim3(gdf), 256, 0, stream>>>(Wf1, wf1t, D, DFF);
  transpose_f16_k<<<dim3(gdf), 256, 0, stream>>>(Wf2, wf2t, DFF, D);

  // --- projections (WMMA GEMMs) ---
  const dim3 gp(M / 64, D / 64);
  gemm_wmma_k<<<gp, 128, 0, stream>>>(xb, wqt,  bq,  qh,   M, D, D, 2, 0);
  gemm_wmma_k<<<gp, 128, 0, stream>>>(xb, wkt,  bk,  kh,   M, D, D, 2, 0);
  gemm_wmma_k<<<gp, 128, 0, stream>>>(xb, wvt,  bv,  vTh,  M, D, D, 3, 0);
  gemm_wmma_k<<<gp, 128, 0, stream>>>(xb, wqgt, bqg, qgh,  M, D, D, 2, 0);
  gemm_wmma_k<<<gp, 128, 0, stream>>>(xb, wkgt, bkg, kgh,  M, D, D, 2, 0);
  gemm_wmma_k<<<gp, 128, 0, stream>>>(xb, wvgt, bvg, vgTh, M, D, D, 3, 0);

  // --- attention ---
  band_attn_k<<<dim3(B * H * NB * (W / 64)), 128, 0, stream>>>(qh, kh, vTh,
                                                               amask, attn);
  global_attn_k<<<dim3(B * H), 128, 0, stream>>>(qgh, kgh, vgTh, amask, attn);

  // --- output projection + LN1 ---
  gemm_wmma_k<<<gp, 128, 0, stream>>>(attn, wot, bo, tmp, M, D, D, 0, 0);
  add_ln_k<<<dim3(M / 4), 128, 0, stream>>>(x, tmp, ln1g, ln1b, h1, h1h, M);

  // --- FFN + LN2 ---
  gemm_wmma_k<<<dim3(M / 64, DFF / 64), 128, 0, stream>>>(h1h, wf1t, bf1, ff,
                                                          M, DFF, D, 1, 1);
  gemm_wmma_k<<<gp, 128, 0, stream>>>(ff, wf2t, bf2, tmp2, M, D, DFF, 0, 0);
  add_ln_k<<<dim3(M / 4), 128, 0, stream>>>(h1, tmp2, ln2g, ln2b,
                                            (float*)d_out, nullptr, M);
}
